// FlowStep_13262859010066
// MI455X (gfx1250) — matrix operand order repaired and checked by reading, not compiled, not run on hardware
//
#include <hip/hip_runtime.h>
#include <hip/hip_bf16.h>

// Problem constants
#define Bb 64
#define Tt 4096
#define Cc 64
#define CONDD 512
#define Hh 256
#define K1 (Cc/2 + CONDD)          // 544 = 17*32
#define NKS1 (K1/32)               // 17
#define NT1 (Hh/16)                // 16
#define NKS2 (Hh/32)               // 8
#define NT2 (Cc/16)                // 4
#define ROWS 64                    // rows per workgroup
#define LSA_STRIDE 552             // 544 + 8 pad (halves), 16B-aligned stride
#define LSH_STRIDE 264             // 256 + 8 pad (halves)

typedef __attribute__((ext_vector_type(16))) _Float16 v16h;
typedef __attribute__((ext_vector_type(8)))  _Float16 v8h;
typedef __attribute__((ext_vector_type(8)))  float    v8f;
typedef __attribute__((ext_vector_type(4)))  float    v4f;   // native vector for NT loads

static __device__ inline v8f zero8() {
    v8f r;
#pragma unroll
    for (int i = 0; i < 8; ++i) r[i] = 0.0f;
    return r;
}

static __device__ inline v16h cat16(v8h lo, v8h hi) {
    v16h r;
#pragma unroll
    for (int i = 0; i < 8; ++i) { r[i] = lo[i]; r[i + 8] = hi[i]; }
    return r;
}

static __device__ inline v8f wmma_f16(v16h a, v16h b, v8f c) {
    return __builtin_amdgcn_wmma_f32_16x16x32_f16(
        /*neg_a=*/false, a, /*neg_b=*/false, b,
        /*c_mod=*/(short)0, c, /*reuse_a=*/false, /*reuse_b=*/false);
}

// ---------------------------------------------------------------------------
// Pack W (K x N fp32, row-major) into per-fragment f16 B operands in contiguous k order for
// V_WMMA_F32_16X16X32_F16: fragment (ks,nt) covers K[ks*32,+32) x N[nt*16,+16).
// Storage: frag_idx-major, then lane (32B per lane): lane l holds column
// N = nt*16 + (l&15); lanes 0-15 hold K 0..15 (2 per VGPR), lanes 16-31 hold
// K 16..31.
// ---------------------------------------------------------------------------
__global__ void pack_w_kernel(const float* __restrict__ W, _Float16* __restrict__ P,
                              int nks, int nnt, int N) {
    int t = blockIdx.x * blockDim.x + threadIdx.x;
    int f = t >> 5, l = t & 31;
    if (f >= nks * nnt) return;
    int ks = f / nnt, nt = f % nnt;
    int n  = nt * 16 + (l & 15);
    int kb = ks * 32 + ((l >> 4) << 4);
    _Float16* dst = P + ((size_t)f * 32 + l) * 16;
#pragma unroll
    for (int v = 0; v < 8; ++v) {
        int k = kb + 2 * v;
        dst[2 * v]     = (_Float16)W[(size_t)k * N + n];
        dst[2 * v + 1] = (_Float16)W[(size_t)(k + 1) * N + n];
    }
}

// ---------------------------------------------------------------------------
// Main fused kernel: actnorm + permute + concat + GEMM1 + leakyReLU + GEMM2 +
// affine-coupling epilogue.  64 rows per 256-thread block (8 wave32s).
// ---------------------------------------------------------------------------
__global__ __launch_bounds__(256) void flowstep_main_kernel(
    const float* __restrict__ input,  // (B,T,64)
    const float* __restrict__ cond,   // (B,T,512)
    const float* __restrict__ abias,  // (64,)
    const float* __restrict__ alogs,  // (64,)
    const int*   __restrict__ perm,   // (64,)
    const float* __restrict__ b1,     // (256,)
    const float* __restrict__ b2,     // (64,)
    const float* __restrict__ logs2,  // (64,)
    const _Float16* __restrict__ pW1, // packed W1 fragments
    const _Float16* __restrict__ pW2, // packed W2 fragments
    float* __restrict__ out,          // (B,T,64)
    float* __restrict__ partial)      // (4096,) per-block logdet partials
{
    __shared__ __align__(16) _Float16 lsA[ROWS][LSA_STRIDE]; // z1(32) ++ cond(512) as f16
    __shared__ __align__(16) _Float16 lsH[ROWS][LSH_STRIDE]; // hidden activations f16
    __shared__ float z2f[ROWS][33];                          // permuted-actnorm z2 fp32
    __shared__ float pb[Cc], ps[Cc];
    __shared__ int   pidx[Cc];
    __shared__ float s_ld;

    const int tid  = threadIdx.x;
    const int lane = tid & 31;
    const int wv   = tid >> 5;
    const long r0  = (long)blockIdx.x * ROWS;

    if (tid < Cc) {
        int pc = perm[tid];
        pidx[tid] = pc;
        pb[tid]   = abias[pc];
        ps[tid]   = expf(alogs[pc]);
    }
    if (tid == 0) s_ld = 0.0f;
    __syncthreads();

    // ---- Stage A: permuted actnorm z.  z1 -> lsA (f16) + global out, z2 -> z2f
    //      input/out are touched once -> non-temporal.
#pragma unroll
    for (int k = 0; k < 16; ++k) {
        int idx = tid + k * 256;              // 64 rows x 64 ch
        int row = idx >> 6, c = idx & 63;
        float iv = __builtin_nontemporal_load(&input[(r0 + row) * Cc + pidx[c]]);
        float z  = (iv + pb[c]) * ps[c];
        if (c < Cc / 2) {
            lsA[row][c] = (_Float16)z;
            __builtin_nontemporal_store(z, &out[(r0 + row) * Cc + c]); // z1 pass-through
        } else {
            z2f[row][c - Cc / 2] = z;
        }
    }
    // ---- Stage B: speech_cond -> lsA cols [32, 544) as f16 (vectorized, NT)
    {
        const v4f* cp = (const v4f*)(cond + r0 * CONDD);
#pragma unroll
        for (int k = 0; k < 32; ++k) {
            int idx = tid + k * 256;          // 64 rows x 128 float4
            int row = idx >> 7, c4 = idx & 127;
            v4f v = __builtin_nontemporal_load(&cp[(long)row * 128 + c4]);
            int cb = Cc / 2 + c4 * 4;
            lsA[row][cb + 0] = (_Float16)v[0];
            lsA[row][cb + 1] = (_Float16)v[1];
            lsA[row][cb + 2] = (_Float16)v[2];
            lsA[row][cb + 3] = (_Float16)v[3];
        }
    }
    __syncthreads();

    // ---- GEMM1: (64 x 544) x (544 x 256).  Wave wv owns n-tiles {2wv,2wv+1},
    //      all 4 m-tiles.  17 K-steps of 32.  All A-fragments of a k-step are
    //      fetched before the WMMA burst so up to 8 ds_loads stay in flight.
    v8f acc[4][2];
#pragma unroll
    for (int mt = 0; mt < 4; ++mt) { acc[mt][0] = zero8(); acc[mt][1] = zero8(); }
    {
        const int nt0 = wv * 2;
        for (int ks = 0; ks < NKS1; ++ks) {
            const _Float16* bp = pW1 + ((size_t)(ks * NT1 + nt0) * 32 + lane) * 16;
            v16h bf0 = *(const v16h*)bp;
            v16h bf1 = *(const v16h*)(bp + 32 * 16);   // next n-tile fragment
            v16h af[4];
#pragma unroll
            for (int mt = 0; mt < 4; ++mt) {
                const _Float16* ap =
                    &lsA[mt * 16 + (lane & 15)][ks * 32 + ((lane >> 4) << 4)];
                af[mt] = cat16(*(const v8h*)ap, *(const v8h*)(ap + 8));
            }
#pragma unroll
            for (int mt = 0; mt < 4; ++mt) {
                acc[mt][0] = wmma_f16(af[mt], bf0, acc[mt][0]);
                acc[mt][1] = wmma_f16(af[mt], bf1, acc[mt][1]);
            }
        }
    }
    // ---- bias + leaky ReLU -> lsH (f16)
#pragma unroll
    for (int j = 0; j < 2; ++j) {
        int n = (wv * 2 + j) * 16 + (lane & 15);
        float bb = b1[n];
#pragma unroll
        for (int mt = 0; mt < 4; ++mt) {
#pragma unroll
            for (int i = 0; i < 8; ++i) {
                int row = mt * 16 + i + ((lane >> 4) << 3);
                float v = acc[mt][j][i] + bb;
                v = v > 0.0f ? v : 0.01f * v;
                lsH[row][n] = (_Float16)v;
            }
        }
    }
    __syncthreads();

    // ---- GEMM2: (64 x 256) x (256 x 64).  16 output tiles over 8 waves:
    //      wave wv -> m-tile (wv&3), n-tiles {2*(wv>>2), 2*(wv>>2)+1}.
    v8f acc2[2] = { zero8(), zero8() };
    const int mt2 = wv & 3;
    const int ntb = (wv >> 2) * 2;
    for (int ks = 0; ks < NKS2; ++ks) {
        const _Float16* bp = pW2 + ((size_t)(ks * NT2 + ntb) * 32 + lane) * 16;
        v16h bf0 = *(const v16h*)bp;
        v16h bf1 = *(const v16h*)(bp + 32 * 16);
        const _Float16* ap =
            &lsH[mt2 * 16 + (lane & 15)][ks * 32 + ((lane >> 4) << 4)];
        v16h af = cat16(*(const v8h*)ap, *(const v8h*)(ap + 8));
        acc2[0] = wmma_f16(af, bf0, acc2[0]);
        acc2[1] = wmma_f16(af, bf1, acc2[1]);
    }

    // ---- Coupling epilogue: h = (acc+b2)*exp(3*logs2); even cols = shift,
    //      odd cols = scale-arg.  Pair lanes via shfl_xor(1).
    float ldacc = 0.0f;
#pragma unroll
    for (int j = 0; j < 2; ++j) {
        int n = (ntb + j) * 16 + (lane & 15);       // output channel in [0,64)
        float bb = b2[n];
        float e2 = expf(logs2[n] * 3.0f);
#pragma unroll
        for (int i = 0; i < 8; ++i) {
            float t = (acc2[j][i] + bb) * e2;
            float o = __shfl_xor(t, 1, 32);         // partner column value
            if ((n & 1) == 0) {
                float scale = fmaxf(1.0f / (1.0f + expf(-(o + 2.0f))), 1e-6f);
                int c2  = n >> 1;                   // z2 channel in [0,32)
                int row = mt2 * 16 + i + ((lane >> 4) << 3);
                float zv = (z2f[row][c2] + t) * scale;
                __builtin_nontemporal_store(zv, &out[(r0 + row) * Cc + Cc / 2 + c2]);
                ldacc += logf(scale);
            }
        }
    }
    if ((lane & 1) == 0) atomicAdd(&s_ld, ldacc);
    __syncthreads();
    if (tid == 0) partial[blockIdx.x] = s_ld;
}

// ---------------------------------------------------------------------------
// logdet[b] = T * sum(actnorm_logs) + sum of the 64 per-block partials of b.
// ---------------------------------------------------------------------------
__global__ void finalize_logdet_kernel(const float* __restrict__ alogs,
                                       const float* __restrict__ partial,
                                       float* __restrict__ ldout) {
    int b = threadIdx.x;
    if (b >= Bb) return;
    float s = 0.0f;
    for (int c = 0; c < Cc; ++c) s += alogs[c];
    float ld = (float)Tt * s;
    for (int g = 0; g < Tt / ROWS; ++g) ld += partial[b * (Tt / ROWS) + g];
    ldout[b] = ld;
}

extern "C" void kernel_launch(void* const* d_in, const int* in_sizes, int n_in,
                              void* d_out, int out_size, void* d_ws, size_t ws_size,
                              hipStream_t stream) {
    const float* input = (const float*)d_in[0];   // (B,T,64)
    const float* cond  = (const float*)d_in[1];   // (B,T,512)
    const float* abias = (const float*)d_in[2];   // (64,)
    const float* alogs = (const float*)d_in[3];   // (64,)
    const int*   perm  = (const int*)  d_in[4];   // (64,)
    const float* W1    = (const float*)d_in[5];   // (544,256)
    const float* b1    = (const float*)d_in[6];   // (256,)
    const float* W2    = (const float*)d_in[7];   // (256,64)
    const float* b2    = (const float*)d_in[8];   // (64,)
    const float* logs2 = (const float*)d_in[9];   // (64,)

    float* out = (float*)d_out;                       // z_seq (B,T,64)
    float* ldo = out + (size_t)Bb * Tt * Cc;          // logdet (B,)

    // Workspace layout
    char* ws = (char*)d_ws;
    _Float16* pW1 = (_Float16*)ws;                                    // 272*512 halves
    _Float16* pW2 = (_Float16*)(ws + (size_t)NKS1 * NT1 * 512 * 2);   // 32*512 halves
    float* partial = (float*)(ws + (size_t)NKS1 * NT1 * 512 * 2
                                 + (size_t)NKS2 * NT2 * 512 * 2);     // 4096 floats

    // 1) Pack weights into WMMA B-fragment order (f16)
    {
        int frags1 = NKS1 * NT1;                 // 272
        int thr1 = frags1 * 32;
        pack_w_kernel<<<(thr1 + 255) / 256, 256, 0, stream>>>(W1, pW1, NKS1, NT1, Hh);
        int frags2 = NKS2 * NT2;                 // 32
        int thr2 = frags2 * 32;
        pack_w_kernel<<<(thr2 + 255) / 256, 256, 0, stream>>>(W2, pW2, NKS2, NT2, Cc);
    }

    // 2) Fused flow step
    {
        dim3 grid((Bb * Tt) / ROWS);             // 4096 blocks
        dim3 block(256);                         // 8 wave32s
        flowstep_main_kernel<<<grid, block, 0, stream>>>(
            input, cond, abias, alogs, perm, b1, b2, logs2, pW1, pW2, out, partial);
    }

    // 3) Deterministic logdet finalize
    finalize_logdet_kernel<<<1, 64, 0, stream>>>(alogs, partial, ldo);
}
